// FloodDetectionGraph_40140764348924
// MI455X (gfx1250) — compile-verified, matrix operand order checked by reading, not run on hardware
//
#include <hip/hip_runtime.h>
#include <cstdint>
#include <cstddef>

// ---------- types ----------
typedef __bf16 bf16;
typedef __bf16 v16bf __attribute__((ext_vector_type(16)));
typedef __bf16 v8bf  __attribute__((ext_vector_type(8)));
typedef float  v8f   __attribute__((ext_vector_type(8)));

union BFrag { v16bf v; v8bf h[2]; };

#define CH 128

// ---------- tiny utility kernels ----------
__global__ void fd_fill(float* __restrict__ p, float v, int n) {
    int i = blockIdx.x * blockDim.x + threadIdx.x;
    if (i < n) p[i] = v;
}

__global__ void fd_deg_edges(const long long* __restrict__ dst,
                             float* __restrict__ deg, int E) {
    int i = blockIdx.x * blockDim.x + threadIdx.x;
    if (i < E) atomicAdd(&deg[(int)dst[i]], 1.0f);
}

__global__ void fd_rsqrt(const float* __restrict__ deg,
                         float* __restrict__ dis, int n) {
    int i = blockIdx.x * blockDim.x + threadIdx.x;
    if (i < n) dis[i] = rsqrtf(deg[i]);
}

__global__ void fd_f32_to_bf16(const float* __restrict__ src,
                               bf16* __restrict__ dst, int n) {
    int i = blockIdx.x * blockDim.x + threadIdx.x;
    if (i < n) dst[i] = (bf16)src[i];
}

// Pack W [128x128 f32 row-major, K x N] into per-lane B-fragment order:
// Bp[((kt*8 + nt)*32 + lane)*16 + i]  (kt: K-tile of 32, nt: N-tile of 16)
__global__ void fd_pack_w(const float* __restrict__ W, bf16* __restrict__ Bp) {
    int idx = blockIdx.x * blockDim.x + threadIdx.x;   // 4*8*32*16 = 16384
    if (idx >= 16384) return;
    int i    = idx & 15;
    int lane = (idx >> 4) & 31;
    int nt   = (idx >> 9) & 7;
    int kt   = (idx >> 12) & 3;
    int n    = lane & 15;
    int kb   = (lane < 16) ? 0 : 8;
    int kk   = (i < 8) ? (kb + i) : (kb + 16 + (i - 8));
    int k    = kt * 32 + kk;
    Bp[idx] = (bf16)W[k * CH + nt * 16 + n];
}

// ---------- WMMA GEMM: C[N,128] = A[N,128](bf16) x W(packed bf16) ----------
// One wave per 16x128 output strip: A fragment loaded ONCE per K-tile,
// reused across 8 N-tile accumulators -> 32 wmma per wave, 8x less A traffic.
__global__ void fd_gemm_wmma(const bf16* __restrict__ A,
                             const bf16* __restrict__ Bp,
                             float* __restrict__ C, int rowTiles) {
    int wave = blockIdx.x * (blockDim.x >> 5) + (threadIdx.x >> 5);
    if (wave >= rowTiles) return;                   // wave-uniform exit
    int lane  = threadIdx.x & 31;

    int row   = wave * 16 + (lane & 15);
    int kbase = (lane < 16) ? 0 : 8;                // A-fragment ISA layout

    v8f acc[8] = {{}, {}, {}, {}, {}, {}, {}, {}};

#pragma unroll
    for (int kt = 0; kt < 4; ++kt) {
        const bf16* ap = A + (size_t)row * CH + kt * 32 + kbase;
        BFrag a;
        a.h[0] = *(const v8bf*)(ap);                // K kbase..kbase+7
        a.h[1] = *(const v8bf*)(ap + 16);           // K kbase+16..kbase+23
        const bf16* bk = Bp + ((size_t)(kt * 8) * 32 << 4) + ((size_t)lane << 4);
#pragma unroll
        for (int nt = 0; nt < 8; ++nt) {
            BFrag b;
            const bf16* bp = bk + ((size_t)(nt * 32) << 4);
            b.h[0] = *(const v8bf*)(bp);
            b.h[1] = *(const v8bf*)(bp + 8);
            acc[nt] = __builtin_amdgcn_wmma_f32_16x16x32_bf16(
                false, a.v, false, b.v, (short)0, acc[nt], false, false);
        }
    }

    // D layout: lane<16 -> M = r, lane>=16 -> M = r+8 ; N = lane&15
    int col16 = lane & 15;
    int rbase = wave * 16 + ((lane < 16) ? 0 : 8);
#pragma unroll
    for (int nt = 0; nt < 8; ++nt) {
        float* cp = C + (size_t)rbase * CH + nt * 16 + col16;
#pragma unroll
        for (int r = 0; r < 8; ++r)
            cp[(size_t)r * CH] = acc[nt][r];
    }
}

// ---------- edge scatter: agg[dst] += hW[src] * dis[src]*dis[dst] ----------
// One wave per edge; lane moves float4 (coalesced 512B/wave), 4 f32 atomics.
__global__ void fd_scatter(const long long* __restrict__ src,
                           const long long* __restrict__ dst,
                           const float* __restrict__ dis,
                           const float* __restrict__ hW,
                           float* __restrict__ agg, int E) {
    int e = blockIdx.x * (blockDim.x >> 5) + (threadIdx.x >> 5);
    if (e >= E) return;
    int lane = threadIdx.x & 31;
    int s = __builtin_amdgcn_readfirstlane((int)src[e]);   // wave-uniform -> SGPR
    int d = __builtin_amdgcn_readfirstlane((int)dst[e]);
    float nrm = dis[s] * dis[d];
    const float4 v = *(const float4*)(hW + (size_t)s * CH + lane * 4);
    float* o = agg + (size_t)d * CH + lane * 4;
    atomicAdd(o + 0, v.x * nrm);
    atomicAdd(o + 1, v.y * nrm);
    atomicAdd(o + 2, v.z * nrm);
    atomicAdd(o + 3, v.w * nrm);
}

// ---------- epilogue: h = relu(agg + hW/deg + b); write f32 (in-place) + bf16
__global__ void fd_finalize(const float* __restrict__ hW,
                            float* __restrict__ agg_io,
                            const float* __restrict__ dis,
                            const float* __restrict__ bias,
                            bf16* __restrict__ hb, int n) {
    int idx = blockIdx.x * blockDim.x + threadIdx.x;
    if (idx >= n) return;
    int i = idx >> 7;
    int c = idx & (CH - 1);
    float di = dis[i];
    float v  = agg_io[idx] + hW[idx] * di * di + bias[c];   // self-loop folded
    v = fmaxf(v, 0.0f);
    agg_io[idx] = v;
    hb[idx]     = (bf16)v;
}

// ---------- global mean pool (accumulate) ----------
__global__ void fd_pool(const float* __restrict__ h,
                        const long long* __restrict__ batch,
                        float* __restrict__ sums,
                        float* __restrict__ cnts, int N) {
    int node = blockIdx.x * (blockDim.x >> 5) + (threadIdx.x >> 5);
    if (node >= N) return;
    int lane = threadIdx.x & 31;
    int g = __builtin_amdgcn_readfirstlane((int)batch[node]);
    const float4 v = *(const float4*)(h + (size_t)node * CH + lane * 4);
    float* o = sums + (size_t)g * CH + lane * 4;
    atomicAdd(o + 0, v.x);
    atomicAdd(o + 1, v.y);
    atomicAdd(o + 2, v.z);
    atomicAdd(o + 3, v.w);
    if (lane == 0) atomicAdd(&cnts[g], 1.0f);
}

// ---------- head: out[g,o] = (sums[g]/cnt) @ Wlin + blin ----------
__global__ void fd_head(const float* __restrict__ sums,
                        const float* __restrict__ cnts,
                        const float* __restrict__ Wlin,
                        const float* __restrict__ blin,
                        float* __restrict__ out, int G) {
    int idx = threadIdx.x;                 // G*2 = 128 threads
    if (idx >= G * 2) return;
    int g = idx >> 1, o = idx & 1;
    float inv = 1.0f / fmaxf(cnts[g], 1.0f);
    float acc = blin[o];
#pragma unroll 8
    for (int c = 0; c < CH; ++c)
        acc = fmaf(sums[g * CH + c] * inv, Wlin[c * 2 + o], acc);
    out[idx] = acc;
}

// ---------- launch ----------
extern "C" void kernel_launch(void* const* d_in, const int* in_sizes, int n_in,
                              void* d_out, int out_size, void* d_ws, size_t ws_size,
                              hipStream_t stream) {
    const float*     x     = (const float*)d_in[0];
    const long long* ei    = (const long long*)d_in[1];   // int64 [2, E]
    const long long* batch = (const long long*)d_in[2];   // int64 [N]
    const float* W[3] = {(const float*)d_in[3], (const float*)d_in[5], (const float*)d_in[7]};
    const float* b[3] = {(const float*)d_in[4], (const float*)d_in[6], (const float*)d_in[8]};
    const float* Wlin = (const float*)d_in[9];
    const float* blin = (const float*)d_in[10];
    float* out = (float*)d_out;

    const int N = in_sizes[0] / CH;   // 50000
    const int E = in_sizes[1] / 2;    // 1600000
    const int G = 64;
    const int NC = N * CH;

    // carve workspace (256B aligned)
    char* ws = (char*)d_ws;
    auto carve = [&](size_t bytes) -> char* {
        char* p = ws;
        ws += (bytes + 255) & ~(size_t)255;
        return p;
    };
    float* agg  = (float*)carve((size_t)NC * 4);   // agg, later holds h (f32)
    float* hW   = (float*)carve((size_t)NC * 4);
    bf16*  hb   = (bf16*) carve((size_t)NC * 2);
    float* deg  = (float*)carve((size_t)N * 4);
    float* dis  = (float*)carve((size_t)N * 4);
    bf16*  Wp   = (bf16*) carve((size_t)3 * 16384 * 2);
    float* sums = (float*)carve((size_t)G * CH * 4);
    float* cnts = (float*)carve((size_t)G * 4);

    const long long* srcI = ei;
    const long long* dstI = ei + E;

    const int T = 256;
    // degrees (self-loop -> init 1) and dis = rsqrt(deg)
    fd_fill     <<<(N + T - 1) / T, T, 0, stream>>>(deg, 1.0f, N);
    fd_deg_edges<<<(E + T - 1) / T, T, 0, stream>>>(dstI, deg, E);
    fd_rsqrt    <<<(N + T - 1) / T, T, 0, stream>>>(deg, dis, N);

    // x -> bf16, pack all three weight matrices
    fd_f32_to_bf16<<<(NC + T - 1) / T, T, 0, stream>>>(x, hb, NC);
    for (int l = 0; l < 3; ++l)
        fd_pack_w<<<64, 256, 0, stream>>>(W[l], Wp + l * 16384);

    const int rowTiles = N / 16;                     // 3125 strips of 16x128
    for (int l = 0; l < 3; ++l) {
        fd_gemm_wmma<<<(rowTiles + 7) / 8, 256, 0, stream>>>(hb, Wp + l * 16384, hW, rowTiles);
        fd_fill     <<<(NC + T - 1) / T, T, 0, stream>>>(agg, 0.0f, NC);
        fd_scatter  <<<(E + 7) / 8, 256, 0, stream>>>(srcI, dstI, dis, hW, agg, E);
        fd_finalize <<<(NC + T - 1) / T, T, 0, stream>>>(hW, agg, dis, b[l], hb, NC);
    }

    // mean pool + linear head
    fd_fill<<<(G * CH + T - 1) / T, T, 0, stream>>>(sums, 0.0f, G * CH);
    fd_fill<<<1, 64, 0, stream>>>(cnts, 0.0f, G);
    fd_pool<<<(N + 7) / 8, 256, 0, stream>>>(agg, batch, sums, cnts, N);
    fd_head<<<1, 128, 0, stream>>>(sums, cnts, Wlin, blin, out, G);
}
